// VanillaRNN_71047349010479
// MI455X (gfx1250) — matrix operand order repaired and checked by reading, not compiled, not run on hardware
//
#include <hip/hip_runtime.h>
#include <stdint.h>
#include <stddef.h>

// ---------------------------------------------------------------------------
// VanillaRNN for MI455X (gfx1250, wave32, WMMA + async-to-LDS + TDM)
//   Phase 1: xw = x @ W_ih^T + b_h          (bf16 WMMA, LDS double-buffered
//            via GLOBAL_LOAD_ASYNC_TO_LDS_B128, f32 accum)
//   Phase 2: h_t = tanh(xw_t + W_hh @ h)    (persistent 32-WG kernel,
//            W_hh slice TDM-staged into 256KB dynamic LDS, async h staging,
//            per-step device-scope barrier)
//   Phase 3: out = hs @ W_ho^T + W_ho_b     (same WMMA kernel)
// ---------------------------------------------------------------------------

#define SEQ  8192
#define DIN  1024
#define DH   2048
#define DOUT 1024

#define RG   32            // recurrence workgroups (device barrier members)
#define RPW  (DH / RG)     // 64 rows of W_hh per workgroup
#define RT   256           // threads per recurrence workgroup

typedef __attribute__((ext_vector_type(16))) __bf16 v16bf;
typedef __attribute__((ext_vector_type(8)))  float  v8f;
typedef __attribute__((ext_vector_type(4)))  uint32_t v4u;
typedef __attribute__((ext_vector_type(8)))  int      v8i;
typedef __attribute__((ext_vector_type(4)))  int      v4i;

union Frag { uint32_t u[8]; v16bf v; };

__device__ __forceinline__ uint16_t f32_to_bf16(float f) {
  union { float f; uint32_t u; } c; c.f = f;
  uint32_t r = c.u + 0x7FFFu + ((c.u >> 16) & 1u);   // round-to-nearest-even
  return (uint16_t)(r >> 16);
}
__device__ __forceinline__ float bf16lo(uint32_t w) {
  union { uint32_t u; float f; } c; c.u = w << 16; return c.f;
}
__device__ __forceinline__ float bf16hi(uint32_t w) {
  union { uint32_t u; float f; } c; c.u = w & 0xFFFF0000u; return c.f;
}

// ---- CDNA5 async copy: global -> LDS, 16B per lane, tracked by ASYNCcnt ---
__device__ __forceinline__ void async_ld_b128(void* lds_ptr, const void* gptr) {
  uint32_t l = (uint32_t)(uintptr_t)lds_ptr;     // LDS aperture: addr[31:0]
  uint64_t g = (uint64_t)(uintptr_t)gptr;
  asm volatile("global_load_async_to_lds_b128 %0, %1, off"
               :: "v"(l), "v"(g) : "memory");
}
__device__ __forceinline__ void wait_async0() {
  asm volatile("s_wait_asynccnt 0x0" ::: "memory");
}

// ---- CDNA5 TDM: bulk tile copy global -> LDS, tracked by TENSORcnt -------
// Copies `bytes` (multiple of 8) from gptr to lds_ptr as a 1-row tile of
// 8-byte elements. D# packing per ISA 8.3/8.4 (count=1, type=2, data_size=3).
// This toolchain exposes the 6-arg builtin (g0, g1, g2, g3, g?, cpol).
__device__ __forceinline__ void tdm_load_to_lds(void* lds_ptr, const void* gptr,
                                                uint32_t bytes) {
  const uint32_t units = bytes >> 3;             // 8-byte elements (<= 0xFFFF)
  const uint64_t ga = (uint64_t)(uintptr_t)gptr;
  v4u g0;
  g0[0] = 1u;                                    // count=1, user descriptor
  g0[1] = (uint32_t)(uintptr_t)lds_ptr;          // lds_addr
  g0[2] = (uint32_t)ga;                          // global_addr[31:0]
  g0[3] = ((uint32_t)(ga >> 32) & 0x01FFFFFFu)   // global_addr[56:32]
          | 0x80000000u;                         // type=2 ("image")
  v8i g1;
  g1[0] = (int)(3u << 16);                       // wg_mask=0, data_size=3 (8B)
  g1[1] = (int)(units << 16);                    // tensor_dim0[15:0] @ bits 63:48
  g1[2] = (int)((units >> 16) | (1u << 16));     // tensor_dim0[31:16], tensor_dim1=1
  g1[3] = (int)(units << 16);                    // tile_dim0 = units
  g1[4] = 1;                                     // tile_dim1 = 1, tile_dim2 = 0
  g1[5] = (int)units;                            // tensor_dim0_stride[31:0]
  g1[6] = 0;                                     // stride0[47:32], stride1[15:0]
  g1[7] = 0;                                     // tensor_dim1_stride[47:16]
  v4i gz4  = {0, 0, 0, 0};
  v8i gz8  = {0, 0, 0, 0, 0, 0, 0, 0};
  __builtin_amdgcn_tensor_load_to_lds(g0, g1, gz4, gz4, gz8, 0);
}
__device__ __forceinline__ void wait_tensor0() {
  __builtin_amdgcn_s_wait_tensorcnt(0);
}

// --------------------------- f32 -> bf16 convert ---------------------------
__global__ void cvt_f32_bf16(const float* __restrict__ src,
                             uint16_t* __restrict__ dst, int n) {
  int i = blockIdx.x * blockDim.x + threadIdx.x;
  if (i < n) dst[i] = f32_to_bf16(src[i]);
}

// ------------------------- per-call state init -----------------------------
__global__ void rnn_init(const float* __restrict__ h0,
                         float* __restrict__ hbuf,
                         int* __restrict__ counters) {
  int i = blockIdx.x * blockDim.x + threadIdx.x;
  if (i < DH) { hbuf[i] = h0[i]; hbuf[DH + i] = 0.0f; }
  if (i < SEQ) counters[i] = 0;
}

// --------------------- bf16 WMMA GEMM:  C = A @ B^T + bias -----------------
// A: M x K bf16 row-major, B: N x K bf16 row-major, C: M x N f32.
// Block (256 thr, 8 waves) computes a 128x64 C tile. Per 32-K step the block
// stages A(128x32) + B(64x32) bf16 into LDS with async-to-LDS, double
// buffered; wave w computes rows w*16..w*16+15 x all 64 cols (4 WMMA tiles).
// Fragment packing: contiguous k = 16*half + s for both A and B (16-bit A 16x32, B 32x16, f32 C).
__global__ void wmma_gemm_bf16(const uint16_t* __restrict__ A,
                               const uint16_t* __restrict__ B,
                               const float* __restrict__ bias,
                               float* __restrict__ C,
                               int M, int N, int K) {
  __shared__ __align__(16) uint8_t tiles[2][12288];   // [A 8KB | B 4KB] x2

  const int tid  = threadIdx.x;
  const int lane = tid & 31;
  const int wave = tid >> 5;
  const int m    = lane & 15;
  const int half = lane >> 4;

  const int nBlk  = N >> 6;
  const int s_blk = (blockIdx.x / nBlk) << 7;   // 128-row tile base
  const int j_blk = (blockIdx.x % nBlk) << 6;   // 64-col tile base

  const uint8_t* Ab = (const uint8_t*)A;
  const uint8_t* Bb = (const uint8_t*)B;

  // Stage one 32-wide K slab into tiles[buf]: 3 async b128 per thread.
  auto stage = [&](int buf, int kt) {
    uint8_t* at = tiles[buf];
    uint8_t* bt = tiles[buf] + 8192;
    int c0 = tid, c1 = tid + 256;
    async_ld_b128(at + c0 * 16,
                  Ab + ((size_t)(s_blk + (c0 >> 2)) * K + kt) * 2 + (c0 & 3) * 16);
    async_ld_b128(at + c1 * 16,
                  Ab + ((size_t)(s_blk + (c1 >> 2)) * K + kt) * 2 + (c1 & 3) * 16);
    async_ld_b128(bt + tid * 16,
                  Bb + ((size_t)(j_blk + (tid >> 2)) * K + kt) * 2 + (tid & 3) * 16);
  };

  v8f acc0 = {}, acc1 = {}, acc2 = {}, acc3 = {};

  const int nk = K >> 5;
  stage(0, 0);
  for (int ik = 0; ik < nk; ++ik) {
    wait_async0();          // this wave's slab chunks landed in LDS
    __syncthreads();        // all waves' chunks landed; prev compute done
    if (ik + 1 < nk) stage((ik + 1) & 1, (ik + 1) << 5);

    const uint32_t* at = (const uint32_t*)(tiles[ik & 1]);
    const uint32_t* bt = (const uint32_t*)(tiles[ik & 1] + 8192);

    // Load A fragment + all 4 B fragments first, then issue WMMAs
    // back-to-back (one dscnt wait, keeps the matrix pipe busy).
    Frag a, b0, b1, b2, b3;
    const uint32_t* arow = at + ((wave << 4) + m) * 16;
#pragma unroll
    for (int i = 0; i < 4; ++i) {
      a.u[i]     = arow[half * 8 + i];
      a.u[4 + i] = arow[half * 8 + 4 + i];
    }
    const uint32_t* brow = bt + m * 16 + half * 8;
#pragma unroll
    for (int i = 0; i < 8; ++i) b0.u[i] = brow[i];
#pragma unroll
    for (int i = 0; i < 8; ++i) b1.u[i] = brow[256 + i];   // +16 rows
#pragma unroll
    for (int i = 0; i < 8; ++i) b2.u[i] = brow[512 + i];   // +32 rows
#pragma unroll
    for (int i = 0; i < 8; ++i) b3.u[i] = brow[768 + i];   // +48 rows

    acc0 = __builtin_amdgcn_wmma_f32_16x16x32_bf16(false, a.v, false, b0.v,
                                                   (short)0, acc0, false, false);
    acc1 = __builtin_amdgcn_wmma_f32_16x16x32_bf16(false, a.v, false, b1.v,
                                                   (short)0, acc1, false, false);
    acc2 = __builtin_amdgcn_wmma_f32_16x16x32_bf16(false, a.v, false, b2.v,
                                                   (short)0, acc2, false, false);
    acc3 = __builtin_amdgcn_wmma_f32_16x16x32_bf16(false, a.v, false, b3.v,
                                                   (short)0, acc3, false, false);
  }

  // C layout: element e -> row s0 + half*8 + e, column j_blk + c*16 + m.
  const int s0 = s_blk + (wave << 4);
  v8f accs[4] = {acc0, acc1, acc2, acc3};
#pragma unroll
  for (int c = 0; c < 4; ++c) {
    const int n = j_blk + (c << 4) + m;
    const float bv = bias[n];
#pragma unroll
    for (int e = 0; e < 8; ++e) {
      const int row = s0 + half * 8 + e;
      C[(size_t)row * N + n] = accs[c][e] + bv;
    }
  }
}

// ------------------------------ recurrence ---------------------------------
// 32 persistent workgroups; WG g owns rows [g*64, g*64+64) of W_hh, staged
// once into 256KB dynamic LDS by the Tensor Data Mover. Each step: async-
// stage h (8KB) into LDS, 4 lanes per row do 512 MACs each, shfl-reduce,
// tanh, publish, device-scope counter barrier.
__global__ void rnn_recurrence(const uint16_t* __restrict__ whh_bf,
                               const float* __restrict__ xw,
                               float* __restrict__ hbuf,        // 2*DH f32
                               uint16_t* __restrict__ hs_bf,    // SEQ*DH bf16
                               float* __restrict__ hfinal_out,  // DH f32
                               int* __restrict__ counters) {    // SEQ ints
  extern __shared__ __align__(16) char smem[];
  float*    h_lds = (float*)smem;                           // DH f32 (8KB)
  uint32_t* w_lds = (uint32_t*)(smem + DH * sizeof(float)); // RPW*DH bf16

  const int tid = threadIdx.x;
  const int r0  = blockIdx.x * RPW;

  // TDM-stage the owned (contiguous) 256KB W_hh slice into LDS: one
  // descriptor, issued by wave 0, waited on TENSORcnt.
  if (tid < 32) {
    tdm_load_to_lds(w_lds, whh_bf + (size_t)r0 * DH, RPW * DH * 2);
    wait_tensor0();
  }
  __syncthreads();

  const int row   = tid >> 2;            // 0..63
  const int q     = tid & 3;             // quarter of K
  const int kbase = q * (DH / 4);        // 512-element span
  const uint32_t* wrow = w_lds + row * (DH / 2) + (kbase >> 1);

  for (int t = 0; t < SEQ; ++t) {
    // Async-stage h_{t-1} (8KB = 512 x 16B chunks) into LDS.
    const uint8_t* hprev = (const uint8_t*)(hbuf + (size_t)(t & 1) * DH);
    uint8_t* hdst = (uint8_t*)h_lds;
    async_ld_b128(hdst + tid * 16,         hprev + tid * 16);
    async_ld_b128(hdst + (tid + 256) * 16, hprev + (tid + 256) * 16);
    wait_async0();
    __syncthreads();

    // 512 MACs per lane from LDS (bf16 weights, f32 h), two accumulators.
    float a0 = 0.0f, a1 = 0.0f;
    const float* hk = h_lds + kbase;
#pragma unroll 8
    for (int d = 0; d < DH / 8; ++d) {   // 256 dwords = 512 elements
      const uint32_t w2 = wrow[d];
      a0 += bf16lo(w2) * hk[2 * d];
      a1 += bf16hi(w2) * hk[2 * d + 1];
    }
    float acc = a0 + a1;
    acc += __shfl_xor(acc, 1);
    acc += __shfl_xor(acc, 2);
    __syncthreads();                     // h_lds reads complete before reuse

    if (q == 0) {
      const float v = xw[(size_t)t * DH + r0 + row] + acc;
      const float h = tanhf(v);
      hbuf[(size_t)((t + 1) & 1) * DH + r0 + row] = h;
      hs_bf[(size_t)t * DH + r0 + row] = f32_to_bf16(h);
      if (t == SEQ - 1) hfinal_out[r0 + row] = h;
    }
    __threadfence();                     // device-scope visibility of h
    __syncthreads();
    if (tid == 0) {
      __hip_atomic_fetch_add(&counters[t], 1, __ATOMIC_ACQ_REL,
                             __HIP_MEMORY_SCOPE_AGENT);
      while (__hip_atomic_load(&counters[t], __ATOMIC_ACQUIRE,
                               __HIP_MEMORY_SCOPE_AGENT) < RG)
        __builtin_amdgcn_s_sleep(1);
    }
    __syncthreads();
  }
}

// ------------------------------- launcher ----------------------------------
extern "C" void kernel_launch(void* const* d_in, const int* in_sizes, int n_in,
                              void* d_out, int out_size, void* d_ws, size_t ws_size,
                              hipStream_t stream) {
  (void)in_sizes; (void)n_in; (void)out_size; (void)ws_size;

  const float* x_seq  = (const float*)d_in[0];  // SEQ*DIN
  const float* h0     = (const float*)d_in[1];  // DH
  const float* W_ih   = (const float*)d_in[2];  // DH*DIN
  const float* W_hh   = (const float*)d_in[3];  // DH*DH
  const float* b_h    = (const float*)d_in[4];  // DH
  const float* W_ho   = (const float*)d_in[5];  // DOUT*DH
  const float* W_ho_b = (const float*)d_in[6];  // DOUT

  // Workspace layout (bytes)
  char* ws = (char*)d_ws;
  uint16_t* xbf    = (uint16_t*)(ws);                                     // 16 MB
  uint16_t* wih_bf = (uint16_t*)(ws + (size_t)16 * 1024 * 1024);          //  4 MB
  uint16_t* whh_bf = (uint16_t*)(ws + (size_t)20 * 1024 * 1024);          //  8 MB
  uint16_t* who_bf = (uint16_t*)(ws + (size_t)28 * 1024 * 1024);          //  4 MB
  float*    xw     = (float*)   (ws + (size_t)32 * 1024 * 1024);          // 64 MB
  uint16_t* hs_bf  = (uint16_t*)(ws + (size_t)96 * 1024 * 1024);          // 32 MB
  float*    hbuf   = (float*)   (ws + (size_t)128 * 1024 * 1024);         // 16 KB
  int*      cnts   = (int*)     (ws + (size_t)128 * 1024 * 1024 + 65536); // 32 KB

  float* outp   = (float*)d_out;             // SEQ*DOUT
  float* hfinal = outp + (size_t)SEQ * DOUT; // DH

  // Per-call state init (graph-replay safe: deterministic every call).
  rnn_init<<<(SEQ + 255) / 256, 256, 0, stream>>>(h0, hbuf, cnts);

  // f32 -> bf16 conversions.
  cvt_f32_bf16<<<(SEQ * DIN + 255) / 256, 256, 0, stream>>>(x_seq, xbf, SEQ * DIN);
  cvt_f32_bf16<<<(DH * DIN + 255) / 256, 256, 0, stream>>>(W_ih, wih_bf, DH * DIN);
  cvt_f32_bf16<<<(DH * DH + 255) / 256, 256, 0, stream>>>(W_hh, whh_bf, DH * DH);
  cvt_f32_bf16<<<(DOUT * DH + 255) / 256, 256, 0, stream>>>(W_ho, who_bf, DOUT * DH);

  // Phase 1: xw = x @ W_ih^T + b_h   (M=SEQ, N=DH, K=DIN)
  {
    const int blocks = (SEQ / 128) * (DH / 64);    // 2048
    wmma_gemm_bf16<<<blocks, 256, 0, stream>>>(xbf, wih_bf, b_h, xw,
                                               SEQ, DH, DIN);
  }

  // Phase 2: sequential recurrence (persistent, device-wide barrier).
  {
    const size_t lds_bytes = (size_t)DH * 4 + (size_t)RPW * DH * 2; // 264 KB
    rnn_recurrence<<<RG, RT, lds_bytes, stream>>>(whh_bf, xw, hbuf, hs_bf,
                                                  hfinal, cnts);
  }

  // Phase 3: out = hs @ W_ho^T + W_ho_b   (M=SEQ, N=DOUT, K=DH)
  {
    const int blocks = (SEQ / 128) * (DOUT / 64);  // 1024
    wmma_gemm_bf16<<<blocks, 256, 0, stream>>>(hs_bf, who_bf, W_ho_b, outp,
                                               SEQ, DOUT, DH);
  }
}
